// CausalAttention_6098853560992
// MI455X (gfx1250) — compile-verified, matrix operand order checked by reading, not run on hardware
//
#include <hip/hip_runtime.h>
#include <math.h>

typedef __attribute__((ext_vector_type(16))) _Float16 v16h;
typedef __attribute__((ext_vector_type(8)))  float    v8f;
typedef __attribute__((ext_vector_type(4)))  unsigned int v4u;
typedef __attribute__((ext_vector_type(4)))  int v4i;
typedef __attribute__((ext_vector_type(8)))  int v8i;

constexpr int BATCH  = 8;
constexpr int SEQ    = 2048;
constexpr int DMODEL = 1024;

// Gather a 16x32 f16 fragment (A layout; B layout is the mirror with lane=N)
// from a row-major source with `strideElems` f16 per row.
// Per ISA 7.12.2 (16-bit A 16x32): lanes 0-15 -> rows, K pairs {0..7,16..23};
// lanes 16-31 -> same rows, K pairs {8..15,24..31}. Pairs read as dwords.
__device__ __forceinline__ v16h load_frag16(const _Float16* __restrict__ p,
                                            int strideElems) {
  const int lane = threadIdx.x & 31;
  const int half = lane >> 4;
  const int row  = lane & 15;
  const unsigned* ru = (const unsigned*)(p + (size_t)row * strideElems);
  const int b0 = half * 4;       // dword index of K base (0 or 8)/2
  const int b1 = 8 + half * 4;   // dword index of K base (16 or 24)/2
  union { unsigned u[8]; v16h v; } f;
#pragma unroll
  for (int g = 0; g < 4; ++g) { f.u[g] = ru[b0 + g]; f.u[4 + g] = ru[b1 + g]; }
  return f.v;
}

__device__ __forceinline__ v8f wmma16x16x32(v16h a, v16h b, v8f c) {
  return __builtin_amdgcn_wmma_f32_16x16x32_f16(false, a, false, b,
                                                (short)0, c, false, false);
}

// Tensor Data Mover: DMA a [rows x cols] f16 tile (row stride `rowStride`
// elements) from global into LDS at byte offset ldsOff. D# packing per
// CDNA5 ISA (2D tile, data_size=2B, type=2).
__device__ __forceinline__ void tdm_load_tile_f16(const _Float16* gsrc,
                                                  unsigned ldsOff, int rows,
                                                  int cols, int rowStride) {
  unsigned long long ga = (unsigned long long)(size_t)gsrc;
  v4u g0;
  g0.x = 1u;                                     // count=1, user mode
  g0.y = ldsOff;                                 // lds_addr (bytes)
  g0.z = (unsigned)(ga & 0xFFFFFFFFu);           // global_addr[31:0]
  g0.w = (unsigned)((ga >> 32) & 0x1FFFFFFu) | (2u << 30);  // [56:32]|type=2
  const unsigned td0 = 1u << 20, td1 = 1u << 20; // tensor dims (no OOB clip)
  v8i g1;
  g1[0] = (int)(1u << 16);                       // data_size=1 (2 bytes)
  g1[1] = (int)((td0 & 0xFFFFu) << 16);          // tensor_dim0 lo16 @ [63:48]
  g1[2] = (int)((td0 >> 16) | ((td1 & 0xFFFFu) << 16));   // dim0 hi / dim1 lo
  g1[3] = (int)((td1 >> 16) | ((unsigned)cols << 16));    // dim1 hi / tile_dim0
  g1[4] = rows;                                  // tile_dim1 (tile_dim2 = 0)
  g1[5] = rowStride;                             // tensor_dim0_stride lo32
  g1[6] = 0;                                     // stride hi / dim1_stride lo
  g1[7] = 0;
  v4i z4 = {0, 0, 0, 0};
#if __clang_major__ >= 23
  v8i z8 = {0, 0, 0, 0, 0, 0, 0, 0};
  __builtin_amdgcn_tensor_load_to_lds(g0, g1, z4, z4, z8, 0);
#else
  __builtin_amdgcn_tensor_load_to_lds(g0, g1, z4, z4, 0);
#endif
}

// Y = X(f32)[SEQ x DMODEL] * W(f32)[DMODEL x DMODEL] -> f16 (optionally Y^T).
template <int TRANS>
__global__ void __launch_bounds__(128)
qkv_gemm_kernel(const float* __restrict__ X, const float* __restrict__ W,
                _Float16* __restrict__ Y, float scale) {
  __shared__ _Float16 lA[64 * 32];                      // X tile   [row][k]
  __shared__ _Float16 lB[64 * 32];                      // W^T tile [n][k]
  __shared__ __align__(16) _Float16 lC[64 * 64];        // C staging
  const int tid  = threadIdx.x;
  const int wave = tid >> 5;
  const int m0 = blockIdx.x * 64;
  const int n0 = blockIdx.y * 64;
  const int wm = (wave >> 1) * 32;
  const int wn = (wave & 1) * 32;

  v8f c[2][2];
#pragma unroll
  for (int i = 0; i < 2; ++i)
#pragma unroll
    for (int j = 0; j < 2; ++j)
#pragma unroll
      for (int r = 0; r < 8; ++r) c[i][j][r] = 0.0f;

  for (int kk = 0; kk < DMODEL; kk += 32) {
#pragma unroll
    for (int it = 0; it < 16; ++it) {                   // stage X (f32->f16)
      int e = tid + 128 * it;
      int row = e >> 5, col = e & 31;
      lA[e] = (_Float16)X[(size_t)(m0 + row) * DMODEL + kk + col];
    }
#pragma unroll
    for (int it = 0; it < 16; ++it) {                   // stage W^T
      int e = tid + 128 * it;
      int n = e >> 5, k = e & 31;
      lB[e] = (_Float16)W[(size_t)(kk + k) * DMODEL + n0 + n];
    }
    __syncthreads();
    v16h a0 = load_frag16(&lA[(wm +  0) * 32], 32);
    v16h a1 = load_frag16(&lA[(wm + 16) * 32], 32);
    v16h b0 = load_frag16(&lB[(wn +  0) * 32], 32);
    v16h b1 = load_frag16(&lB[(wn + 16) * 32], 32);
    c[0][0] = wmma16x16x32(a0, b0, c[0][0]);
    c[0][1] = wmma16x16x32(a0, b1, c[0][1]);
    c[1][0] = wmma16x16x32(a1, b0, c[1][0]);
    c[1][1] = wmma16x16x32(a1, b1, c[1][1]);
    __syncthreads();
  }

  // Epilogue: C-layout -> LDS (transpose there if needed) -> b128 stores.
  const int lane = tid & 31;
  const int cn = lane & 15;
  const int rb = (lane >> 4) * 8;
#pragma unroll
  for (int i = 0; i < 2; ++i)
#pragma unroll
    for (int j = 0; j < 2; ++j)
#pragma unroll
      for (int r = 0; r < 8; ++r) {
        int ml = wm + i * 16 + rb + r;
        int nl = wn + j * 16 + cn;
        _Float16 v = (_Float16)(c[i][j][r] * scale);
        if (TRANS) lC[nl * 64 + ml] = v;
        else       lC[ml * 64 + nl] = v;
      }
  __syncthreads();
  const uint4* ls = (const uint4*)lC;                   // 512 x 16B
#pragma unroll
  for (int t = 0; t < 4; ++t) {
    int idx  = tid + 128 * t;
    int row  = idx >> 3;                                // local row (m or n)
    int col8 = (idx & 7) * 8;                           // f16 column
    uint4 v = ls[idx];
    if (TRANS) *(uint4*)&Y[(size_t)(n0 + row) * SEQ    + m0 + col8] = v;
    else       *(uint4*)&Y[(size_t)(m0 + row) * DMODEL + n0 + col8] = v;
  }
}

// Flash attention, causal. One wave: 16 query rows x 128 output columns.
// K tiles are DMA'd into LDS by the Tensor Data Mover; Q pre-scaled by
// 1/sqrt(D); Vt is [DMODEL][SEQ].
__global__ void __launch_bounds__(128)
attn_kernel(const _Float16* __restrict__ Q, const _Float16* __restrict__ K,
            const _Float16* __restrict__ Vt, float* __restrict__ O) {
  __shared__ _Float16 ldsK[32 * 512];      // 32 keys x 512 d-slice (32 KB)
  __shared__ _Float16 lP[4][16 * 32];      // wave-private P staging
  const int tid = threadIdx.x, wave = tid >> 5, lane = tid & 31;
  const int qr0 = blockIdx.x * 64 + wave * 16;
  const int cb  = blockIdx.y * 128;
  const int cn = lane & 15;
  const int rb = (lane >> 4) * 8;
  _Float16* myP = lP[wave];

  v8f acc[8];
  float m[8], l[8];
#pragma unroll
  for (int r = 0; r < 8; ++r) {
    m[r] = -__builtin_inff();
    l[r] = 0.0f;
#pragma unroll
    for (int ct = 0; ct < 8; ++ct) acc[ct][r] = 0.0f;
  }

  const int jend = blockIdx.x * 64 + 64;   // causal key bound (exclusive)
  for (int j = 0; j < jend; j += 32) {
    __builtin_prefetch(Vt + (size_t)cb * SEQ + j, 0, 1);

    // ---- scores S = Q K^T: contract DMODEL in two TDM-staged 512 slices ----
    v8f s0, s1;
#pragma unroll
    for (int r = 0; r < 8; ++r) { s0[r] = 0.0f; s1[r] = 0.0f; }
    for (int dh = 0; dh < DMODEL; dh += 512) {
      __syncthreads();                     // all readers of ldsK done
      if (wave == 0) {
        tdm_load_tile_f16(K + (size_t)j * DMODEL + dh,
                          (unsigned)(size_t)(void*)ldsK, 32, 512, DMODEL);
        __builtin_amdgcn_s_wait_tensorcnt(0);
      }
      __syncthreads();                     // DMA complete, data visible
#pragma unroll 4
      for (int d0 = 0; d0 < 512; d0 += 32) {
        v16h qa = load_frag16(Q + (size_t)qr0 * DMODEL + dh + d0, DMODEL);
        v16h k0 = load_frag16(&ldsK[d0], 512);
        v16h k1 = load_frag16(&ldsK[16 * 512 + d0], 512);
        s0 = wmma16x16x32(qa, k0, s0);
        s1 = wmma16x16x32(qa, k1, s1);
      }
    }

    // ---- causal mask + online softmax (rows live across 16-lane halves) ----
#pragma unroll
    for (int r = 0; r < 8; ++r) {
      int q = qr0 + rb + r;
      float x0 = (j + cn      <= q) ? s0[r] : -__builtin_inff();
      float x1 = (j + 16 + cn <= q) ? s1[r] : -__builtin_inff();
      float t = fmaxf(x0, x1);
      t = fmaxf(t, __shfl_xor(t, 1, 32));
      t = fmaxf(t, __shfl_xor(t, 2, 32));
      t = fmaxf(t, __shfl_xor(t, 4, 32));
      t = fmaxf(t, __shfl_xor(t, 8, 32));
      float mn = fmaxf(m[r], t);
      float alpha = __expf(m[r] - mn);
      float p0 = __expf(x0 - mn);
      float p1 = __expf(x1 - mn);
      float sum = p0 + p1;
      sum += __shfl_xor(sum, 1, 32);
      sum += __shfl_xor(sum, 2, 32);
      sum += __shfl_xor(sum, 4, 32);
      sum += __shfl_xor(sum, 8, 32);
      l[r] = l[r] * alpha + sum;
      m[r] = mn;
#pragma unroll
      for (int ct = 0; ct < 8; ++ct) acc[ct][r] *= alpha;
      myP[(rb + r) * 32 + cn]      = (_Float16)p0;   // C layout -> row-major
      myP[(rb + r) * 32 + 16 + cn] = (_Float16)p1;
    }
    // Wave-private LDS RAW: per-wave DS ops are in-order; fence the compiler.
    asm volatile("s_wait_dscnt 0" ::: "memory");

    // ---- O += P * V  (B fragments straight from Vt rows) ----
    v16h pa = load_frag16(myP, 32);
#pragma unroll
    for (int ct = 0; ct < 8; ++ct) {
      v16h vb = load_frag16(Vt + (size_t)(cb + ct * 16) * SEQ + j, SEQ);
      acc[ct] = wmma16x16x32(pa, vb, acc[ct]);
    }
  }

  // ---- epilogue: O / l ----
#pragma unroll
  for (int r = 0; r < 8; ++r) {
    float inv = 1.0f / l[r];
    int q = qr0 + rb + r;
#pragma unroll
    for (int ct = 0; ct < 8; ++ct)
      O[(size_t)q * DMODEL + cb + ct * 16 + cn] = acc[ct][r] * inv;
  }
}

extern "C" void kernel_launch(void* const* d_in, const int* in_sizes, int n_in,
                              void* d_out, int out_size, void* d_ws, size_t ws_size,
                              hipStream_t stream) {
  const float* x  = (const float*)d_in[0];
  const float* Wq = (const float*)d_in[1];
  const float* Wk = (const float*)d_in[2];
  const float* Wv = (const float*)d_in[3];
  float* out = (float*)d_out;

  // Workspace: per-batch Q, K (row-major f16) and V^T (f16). ~12.6 MB.
  _Float16* Qs = (_Float16*)d_ws;
  _Float16* Ks = Qs + (size_t)SEQ * DMODEL;
  _Float16* Vt = Ks + (size_t)SEQ * DMODEL;

  const float qscale = 1.0f / 32.0f;   // 1/sqrt(1024)
  const dim3 ggrid(SEQ / 64, DMODEL / 64);
  const dim3 agrid(SEQ / 64, DMODEL / 128);

  for (int b = 0; b < BATCH; ++b) {
    const float* xb = x + (size_t)b * SEQ * DMODEL;
    float* ob = out + (size_t)b * SEQ * DMODEL;
    qkv_gemm_kernel<0><<<ggrid, 128, 0, stream>>>(xb, Wq, Qs, qscale);
    qkv_gemm_kernel<0><<<ggrid, 128, 0, stream>>>(xb, Wk, Ks, 1.0f);
    qkv_gemm_kernel<1><<<ggrid, 128, 0, stream>>>(xb, Wv, Vt, 1.0f);
    attn_kernel<<<agrid, 128, 0, stream>>>(Qs, Ks, Vt, ob);
  }
}
